// GraphMultiHeadSelfAttention_65171833750106
// MI455X (gfx1250) — matrix operand order repaired and checked by reading, not compiled, not run on hardware
//
#include <hip/hip_runtime.h>

#define Bq   4
#define Sq   512
#define Dq   768
#define Hq   12
#define HDq  64
#define NRELq 101
#define NRP  112      // padded relation count (7 * 16)
#define NRP2 128      // padded relation count for K-dim of WMMA (4 * 32)
#define Mrows (Bq*Sq) // 2048
#define KT   (Dq/32)  // 24 k-steps for D-wide GEMMs

typedef __attribute__((ext_vector_type(16))) _Float16 v16h;
typedef __attribute__((ext_vector_type(8)))  float    v8f;
typedef __attribute__((ext_vector_type(4)))  int      v4i;

#if defined(__has_builtin)
#if __has_builtin(__builtin_amdgcn_global_load_async_to_lds_b128) && \
    __has_builtin(__builtin_amdgcn_s_wait_asynccnt)
#define USE_ASYNC_LDS 1
#endif
#endif
#ifndef USE_ASYNC_LDS
#define USE_ASYNC_LDS 0
#endif

#if USE_ASYNC_LDS
typedef __attribute__((address_space(1))) v4i GV4;
typedef __attribute__((address_space(3))) v4i LV4;
__device__ __forceinline__ void async_ld_b128(const void* g, void* l) {
  __builtin_amdgcn_global_load_async_to_lds_b128((GV4*)(g), (LV4*)(l), 0, 0);
}
#endif

__device__ __forceinline__ v8f wmma32(v16h a, v16h b, v8f c) {
  return __builtin_amdgcn_wmma_f32_16x16x32_f16(false, a, false, b, (short)0, c,
                                                false, false);
}

// A operand: 16x32 f16 tile; p -> element (row 0, k0) of row-major src, ld in elems.
// Lanes 0-15 hold M=lane, K {0..7,16..23}; lanes 16-31 hold M=lane-16, K {8..15,24..31}.
__device__ __forceinline__ v16h load_A16(const _Float16* p, int ld, int lane) {
  int r  = lane & 15;
  int hi = (lane >> 4) & 1;
  const _Float16* q0 = p + (size_t)r * ld + hi * 8;
  const _Float16* q1 = q0 + 16;
  v16h a;
#pragma unroll
  for (int i = 0; i < 8; ++i) a[i] = q0[i];
#pragma unroll
  for (int i = 0; i < 8; ++i) a[8 + i] = q1[i];
  return a;
}

// Same A layout, sourced from f32 (LDS) with conversion.
__device__ __forceinline__ v16h load_A16_f32(const float* p, int ld, int k0, int lane) {
  int r  = lane & 15;
  int hi = (lane >> 4) & 1;
  const float* q0 = p + (size_t)r * ld + k0 + hi * 8;
  const float* q1 = q0 + 16;
  v16h a;
#pragma unroll
  for (int i = 0; i < 8; ++i) a[i] = (_Float16)q0[i];
#pragma unroll
  for (int i = 0; i < 8; ++i) a[8 + i] = (_Float16)q1[i];
  return a;
}

// B operand: 32x16 f16 tile with B(k,n) = rows[n][k]; p -> (row n0, col k0), ld elems.
// Lane n (0-15) holds column n, K {0..7,16..23}; lane n+16 holds column n, K {8..15,24..31}.
__device__ __forceinline__ v16h load_B16(const _Float16* p, int ld, int lane) {
  int c  = lane & 15;
  int hi = (lane >> 4) & 1;
  const _Float16* q = p + (size_t)c * ld + hi * 8;
  v16h b;
#pragma unroll
  for (int i = 0; i < 8; ++i) b[i] = q[i];
#pragma unroll
  for (int i = 0; i < 8; ++i) b[8 + i] = q[16 + i];
  return b;
}

// ---------------- prep kernels ----------------

__global__ __launch_bounds__(256) void cvt_f32_f16(const float* __restrict__ s,
                                                   _Float16* __restrict__ d, int n) {
  int i = blockIdx.x * 256 + threadIdx.x;
  if (i < n) d[i] = (_Float16)s[i];
}

// Ek/Eq (101x64) -> f16 padded to (112x64); rows >= 101 zeroed.
__global__ __launch_bounds__(256) void pad_embed(const float* __restrict__ s,
                                                 _Float16* __restrict__ d) {
  int i = blockIdx.x * 256 + threadIdx.x;
  if (i >= NRP * HDq) return;
  int r = i / HDq, e = i % HDq;
  d[i] = (_Float16)(r < NRELq ? s[r * HDq + e] : 0.f);
}

// Ev (101x64) -> EvT f16 (64x128); cols >= 101 zeroed.
__global__ __launch_bounds__(256) void evt_prep(const float* __restrict__ Ev,
                                                _Float16* __restrict__ d) {
  int i = blockIdx.x * 256 + threadIdx.x;
  if (i >= HDq * NRP2) return;
  int e = i / NRP2, r = i % NRP2;
  d[i] = (_Float16)(r < NRELq ? Ev[r * HDq + e] : 0.f);
}

// ---------------- QKV projection: X @ W^T + b ----------------
// Block = 8 waves computing a 128x64 output tile; the 64x32 weight panel is
// double-buffered through LDS (async global->LDS when available) and shared
// by all 8 waves. Software-pipelined: A(k+1)/Bpanel(k+1) in flight during
// WMMA on step k.
__global__ __launch_bounds__(256) void qkv_gemm(
    const _Float16* __restrict__ Xh,   // [3][M][D]
    const _Float16* __restrict__ Wh,   // [3][D][D]
    const float* __restrict__ bqp, const float* __restrict__ bkp,
    const float* __restrict__ bvp,
    _Float16* __restrict__ QH, _Float16* __restrict__ KH,
    _Float16* __restrict__ Vt) {
  __shared__ __align__(16) _Float16 sB[2][64 * 32];   // 2 x 4 KB weight panels
  int wave = threadIdx.x >> 5, lane = threadIdx.x & 31;
  int which = blockIdx.z;
  int m0 = blockIdx.x * 128 + wave * 16;
  int nstrip = blockIdx.y;
  int n0 = nstrip * 64;
  const _Float16* X = Xh + (size_t)which * Mrows * Dq;
  const _Float16* W = Wh + (size_t)which * Dq * Dq;
  const float* bias = (which == 0) ? bqp : (which == 1) ? bkp : bvp;

  int srow = threadIdx.x >> 2;          // 0..63: local weight row
  int scol = (threadIdx.x & 3) * 8;     // 16B chunk within the 32-elem k-panel
  const _Float16* wsrc = W + (size_t)(n0 + srow) * Dq + scol;
  _Float16* sdst0 = &sB[0][srow * 32 + scol];
  _Float16* sdst1 = &sB[1][srow * 32 + scol];

  // prologue: stage k-step 0 into buffer 0
#if USE_ASYNC_LDS
  async_ld_b128(wsrc, sdst0);
  __builtin_amdgcn_s_wait_asynccnt(0);
#else
  *(int4*)sdst0 = *(const int4*)wsrc;
#endif
  v16h aCur = load_A16(X + (size_t)m0 * Dq, Dq, lane);
  __syncthreads();

  v8f acc[4] = {};
  for (int kt = 0; kt < KT; ++kt) {
    int cur = kt & 1;
    _Float16* sdstN = cur ? sdst0 : sdst1;
    const _Float16* curB = &sB[cur][0];
    v16h a = aCur;
    bool more = (kt + 1 < KT);
#if USE_ASYNC_LDS
    if (more) async_ld_b128(wsrc + (kt + 1) * 32, sdstN);
#else
    int4 bstage = {};
    if (more) bstage = *(const int4*)(wsrc + (kt + 1) * 32);
#endif
    if (more) aCur = load_A16(X + (size_t)m0 * Dq + (kt + 1) * 32, Dq, lane);
#pragma unroll
    for (int t = 0; t < 4; ++t) {
      v16h bm = load_B16(curB + t * 16 * 32, 32, lane);
      acc[t] = wmma32(a, bm, acc[t]);
    }
#if USE_ASYNC_LDS
    __builtin_amdgcn_s_wait_asynccnt(0);
#else
    if (more) *(int4*)sdstN = bstage;
#endif
    __syncthreads();
  }

  int c = lane & 15;
  int rbase = (lane < 16) ? 0 : 8;
#pragma unroll
  for (int t = 0; t < 4; ++t) {
    int n = n0 + t * 16 + c;
    float bn = bias[n];
#pragma unroll
    for (int r = 0; r < 8; ++r) {
      int m = m0 + rbase + r;
      float val = acc[t][r] + bn;
      if (which == 0) {
        QH[(size_t)m * Dq + n] = (_Float16)val;
      } else if (which == 1) {
        KH[(size_t)m * Dq + n] = (_Float16)val;
      } else {
        int b = m >> 9, i = m & (Sq - 1);
        int e = t * 16 + c;
        Vt[((size_t)(b * Hq + nstrip) * HDq + e) * Sq + i] = (_Float16)val;
      }
    }
  }
}

// ---------------- relation score table: rel[b,h,i,r] = q_i.Ek[r] + k_i.Eq[r] ----
__global__ __launch_bounds__(256) void rel_gemm(
    const _Float16* __restrict__ QH, const _Float16* __restrict__ KH,
    const _Float16* __restrict__ EkH, const _Float16* __restrict__ EqH,
    float* __restrict__ rel) {
  int wave = threadIdx.x >> 5, lane = threadIdx.x & 31;
  int sid = blockIdx.x * 8 + wave;            // 0..1535
  int bh = sid >> 5;                          // 0..47
  int it = sid & 31;
  int b = bh / Hq, h = bh % Hq;
  int i0 = it * 16;
  const _Float16* qb = QH + ((size_t)(b * Sq + i0) * Dq + h * HDq);
  const _Float16* kb = KH + ((size_t)(b * Sq + i0) * Dq + h * HDq);
  v16h aq0 = load_A16(qb, Dq, lane);
  v16h aq1 = load_A16(qb + 32, Dq, lane);
  v16h ak0 = load_A16(kb, Dq, lane);
  v16h ak1 = load_A16(kb + 32, Dq, lane);
  int c = lane & 15;
  int rbase = (lane < 16) ? 0 : 8;
  for (int nt = 0; nt < 7; ++nt) {
    v8f acc = {};
    acc = wmma32(aq0, load_B16(EkH + (size_t)(nt * 16) * HDq, HDq, lane), acc);
    acc = wmma32(aq1, load_B16(EkH + (size_t)(nt * 16) * HDq + 32, HDq, lane), acc);
    acc = wmma32(ak0, load_B16(EqH + (size_t)(nt * 16) * HDq, HDq, lane), acc);
    acc = wmma32(ak1, load_B16(EqH + (size_t)(nt * 16) * HDq + 32, HDq, lane), acc);
#pragma unroll
    for (int r = 0; r < 8; ++r) {
      int i = i0 + rbase + r;
      rel[((size_t)bh * Sq + i) * NRP + nt * 16 + c] = acc[r];
    }
  }
}

// ---------------- fused attention: scores + LUT + softmax + context ----------
__global__ __launch_bounds__(256) void attn_kernel(
    const _Float16* __restrict__ QH, const _Float16* __restrict__ KH,
    const _Float16* __restrict__ Vt, const float* __restrict__ rel,
    const long long* __restrict__ arc, const int* __restrict__ mask,
    const _Float16* __restrict__ EvT, _Float16* __restrict__ ctxH) {
  __shared__ float sScores[16][Sq];           // 32 KB
  __shared__ unsigned char sArc[16][Sq];      // 8 KB
  __shared__ float sRel[16][NRP];             // 7 KB
  __shared__ float sWsum[16][NRP2];           // 8 KB
  __shared__ unsigned char sMask[Sq];
  __shared__ float sRed[16][16];
  __shared__ float sRowStat[16];

  const float NEG_INF = -__builtin_inff();
  int tid = threadIdx.x;
  int bid = blockIdx.x;
  int it = bid & 31;
  int bh = bid >> 5;
  int b = bh / Hq, h = bh % Hq;
  int i0 = it * 16;

  // phase 0: stage arc (as u8), rel LUT, mask flags; zero bins
  for (int idx = tid; idx < 16 * Sq; idx += 256) {
    int i = idx >> 9, j = idx & (Sq - 1);
    sArc[i][j] = (unsigned char)arc[(size_t)(b * Sq + i0 + i) * Sq + j];
  }
  for (int idx = tid; idx < 16 * NRP; idx += 256) {
    int i = idx / NRP, r = idx % NRP;
    sRel[i][r] = rel[((size_t)bh * Sq + i0 + i) * NRP + r];
  }
  for (int idx = tid; idx < Sq; idx += 256)
    sMask[idx] = (unsigned char)(mask[b * Sq + idx] != 0);
  for (int idx = tid; idx < 16 * NRP2; idx += 256)
    (&sWsum[0][0])[idx] = 0.f;
  __syncthreads();

  int wave = tid >> 5, lane = tid & 31;
  int cidx = lane & 15;
  int rbase = (lane < 16) ? 0 : 8;

  // phase 1: scores = (Q K^T + rel[arc]) / sqrt(HD), masked
  const _Float16* qb = QH + ((size_t)(b * Sq + i0) * Dq + h * HDq);
  v16h a0 = load_A16(qb, Dq, lane);
  v16h a1 = load_A16(qb + 32, Dq, lane);
  for (int jt = wave; jt < Sq / 16; jt += 8) {    // 4 uniform iterations per wave
    int j0 = jt * 16;
    const _Float16* kb = KH + ((size_t)(b * Sq + j0) * Dq + h * HDq);
    if (jt + 8 < Sq / 16)
      __builtin_prefetch(KH + ((size_t)(b * Sq + (jt + 8) * 16) * Dq + h * HDq), 0, 1);
    v8f acc = {};
    acc = wmma32(a0, load_B16(kb, Dq, lane), acc);
    acc = wmma32(a1, load_B16(kb + 32, Dq, lane), acc);
    int j = j0 + cidx;
    bool ok = sMask[j] != 0;
#pragma unroll
    for (int r = 0; r < 8; ++r) {
      int i = rbase + r;
      float v = acc[r] + sRel[i][sArc[i][j]];
      sScores[i][j] = ok ? v * 0.125f : NEG_INF;
    }
  }
  __syncthreads();

  // phase 2: row softmax + relation-bin accumulation (16 threads per row)
  {
    int row = tid >> 4, c = tid & 15;
    float m = NEG_INF;
    for (int j = c; j < Sq; j += 16) m = fmaxf(m, sScores[row][j]);
    sRed[row][c] = m;
    __syncthreads();
    if (c == 0) {
      float mm = NEG_INF;
      for (int k = 0; k < 16; ++k) mm = fmaxf(mm, sRed[row][k]);
      sRowStat[row] = mm;
    }
    __syncthreads();
    float rmax = sRowStat[row];
    float s = 0.f;
    for (int j = c; j < Sq; j += 16) {
      float p = __expf(sScores[row][j] - rmax);
      sScores[row][j] = p;
      s += p;
    }
    sRed[row][c] = s;
    __syncthreads();
    if (c == 0) {
      float ss = 0.f;
      for (int k = 0; k < 16; ++k) ss += sRed[row][k];
      sRowStat[row] = ss;
    }
    __syncthreads();
    float inv = 1.f / sRowStat[row];
    for (int j = c; j < Sq; j += 16) {
      float w = sScores[row][j] * inv;
      sScores[row][j] = w;
      atomicAdd(&sWsum[row][(int)sArc[row][j]], w);   // ds_add_f32
    }
  }
  __syncthreads();

  // phase 3: ctx = W x V  +  bins x Ev   (waves 0..3, one 16-col tile each)
  if (wave < 4) {
    int e0 = wave * 16;
    v8f acc = {};
    for (int kt = 0; kt < Sq / 32; ++kt) {
      v16h a = load_A16_f32(&sScores[0][0], Sq, kt * 32, lane);
      v16h bm = load_B16(
          Vt + ((size_t)(b * Hq + h) * HDq + e0) * Sq + kt * 32, Sq, lane);
      acc = wmma32(a, bm, acc);
    }
    for (int kt = 0; kt < NRP2 / 32; ++kt) {
      v16h a = load_A16_f32(&sWsum[0][0], NRP2, kt * 32, lane);
      v16h bm = load_B16(EvT + (size_t)e0 * NRP2 + kt * 32, NRP2, lane);
      acc = wmma32(a, bm, acc);
    }
#pragma unroll
    for (int r = 0; r < 8; ++r) {
      int i = i0 + rbase + r;
      ctxH[(size_t)(b * Sq + i) * Dq + h * HDq + e0 + cidx] = (_Float16)acc[r];
    }
  }
}

// ---------------- output projection: ctx @ Wo^T + bo ----------------
// Same LDS double-buffered pipelined structure as qkv_gemm.
__global__ __launch_bounds__(256) void out_gemm(
    const _Float16* __restrict__ ctxH, const _Float16* __restrict__ WoH,
    const float* __restrict__ bo, float* __restrict__ out) {
  __shared__ __align__(16) _Float16 sB[2][64 * 32];
  int wave = threadIdx.x >> 5, lane = threadIdx.x & 31;
  int m0 = blockIdx.x * 128 + wave * 16;
  int n0 = blockIdx.y * 64;

  int srow = threadIdx.x >> 2;
  int scol = (threadIdx.x & 3) * 8;
  const _Float16* wsrc = WoH + (size_t)(n0 + srow) * Dq + scol;
  _Float16* sdst0 = &sB[0][srow * 32 + scol];
  _Float16* sdst1 = &sB[1][srow * 32 + scol];

#if USE_ASYNC_LDS
  async_ld_b128(wsrc, sdst0);
  __builtin_amdgcn_s_wait_asynccnt(0);
#else
  *(int4*)sdst0 = *(const int4*)wsrc;
#endif
  v16h aCur = load_A16(ctxH + (size_t)m0 * Dq, Dq, lane);
  __syncthreads();

  v8f acc[4] = {};
  for (int kt = 0; kt < KT; ++kt) {
    int cur = kt & 1;
    _Float16* sdstN = cur ? sdst0 : sdst1;
    const _Float16* curB = &sB[cur][0];
    v16h a = aCur;
    bool more = (kt + 1 < KT);
#if USE_ASYNC_LDS
    if (more) async_ld_b128(wsrc + (kt + 1) * 32, sdstN);
#else
    int4 bstage = {};
    if (more) bstage = *(const int4*)(wsrc + (kt + 1) * 32);
#endif
    if (more) aCur = load_A16(ctxH + (size_t)m0 * Dq + (kt + 1) * 32, Dq, lane);
#pragma unroll
    for (int t = 0; t < 4; ++t) {
      v16h bm = load_B16(curB + t * 16 * 32, 32, lane);
      acc[t] = wmma32(a, bm, acc[t]);
    }
#if USE_ASYNC_LDS
    __builtin_amdgcn_s_wait_asynccnt(0);
#else
    if (more) *(int4*)sdstN = bstage;
#endif
    __syncthreads();
  }

  int c = lane & 15, rbase = (lane < 16) ? 0 : 8;
#pragma unroll
  for (int t = 0; t < 4; ++t) {
    int n = n0 + t * 16 + c;
    float bn = bo[n];
#pragma unroll
    for (int r = 0; r < 8; ++r) {
      int m = m0 + rbase + r;
      out[(size_t)m * Dq + n] = acc[t][r] + bn;
    }
  }
}

extern "C" void kernel_launch(void* const* d_in, const int* in_sizes, int n_in,
                              void* d_out, int out_size, void* d_ws, size_t ws_size,
                              hipStream_t stream) {
  const float*     query = (const float*)d_in[0];
  const float*     key   = (const float*)d_in[1];
  const float*     value = (const float*)d_in[2];
  const int*       mask  = (const int*)d_in[3];
  const long long* arc   = (const long long*)d_in[4];
  const float* Wq = (const float*)d_in[5];
  const float* bq = (const float*)d_in[6];
  const float* Wk = (const float*)d_in[7];
  const float* bk = (const float*)d_in[8];
  const float* Wv = (const float*)d_in[9];
  const float* bv = (const float*)d_in[10];
  const float* Wo = (const float*)d_in[11];
  const float* bo = (const float*)d_in[12];
  const float* Ek = (const float*)d_in[13];
  const float* Eq = (const float*)d_in[14];
  const float* Ev = (const float*)d_in[15];
  (void)in_sizes; (void)n_in; (void)out_size; (void)ws_size;

  char* ws = (char*)d_ws;
  size_t off = 0;
  auto take = [&](size_t bytes) -> char* {
    char* p = ws + off;
    off += (bytes + 255) & ~(size_t)255;
    return p;
  };
  _Float16* Xh   = (_Float16*)take((size_t)3 * Mrows * Dq * 2);
  _Float16* Wh   = (_Float16*)take((size_t)3 * Dq * Dq * 2);
  _Float16* WoH  = (_Float16*)take((size_t)Dq * Dq * 2);
  _Float16* QH   = (_Float16*)take((size_t)Mrows * Dq * 2);
  _Float16* KH   = (_Float16*)take((size_t)Mrows * Dq * 2);
  _Float16* Vt   = (_Float16*)take((size_t)Mrows * Dq * 2);
  _Float16* ctxH = (_Float16*)take((size_t)Mrows * Dq * 2);
  _Float16* EkH  = (_Float16*)take((size_t)NRP * HDq * 2);
  _Float16* EqH  = (_Float16*)take((size_t)NRP * HDq * 2);
  _Float16* EvT  = (_Float16*)take((size_t)HDq * NRP2 * 2);
  float*    relW = (float*)take((size_t)Bq * Hq * Sq * NRP * 4);

  const int nX = Mrows * Dq;   // 1,572,864 (divisible by 256)
  const int nW = Dq * Dq;      //   589,824 (divisible by 256)
  cvt_f32_f16<<<nX / 256, 256, 0, stream>>>(query, Xh, nX);
  cvt_f32_f16<<<nX / 256, 256, 0, stream>>>(key, Xh + (size_t)nX, nX);
  cvt_f32_f16<<<nX / 256, 256, 0, stream>>>(value, Xh + (size_t)2 * nX, nX);
  cvt_f32_f16<<<nW / 256, 256, 0, stream>>>(Wq, Wh, nW);
  cvt_f32_f16<<<nW / 256, 256, 0, stream>>>(Wk, Wh + (size_t)nW, nW);
  cvt_f32_f16<<<nW / 256, 256, 0, stream>>>(Wv, Wh + (size_t)2 * nW, nW);
  cvt_f32_f16<<<nW / 256, 256, 0, stream>>>(Wo, WoH, nW);
  pad_embed<<<(NRP * HDq + 255) / 256, 256, 0, stream>>>(Ek, EkH);
  pad_embed<<<(NRP * HDq + 255) / 256, 256, 0, stream>>>(Eq, EqH);
  evt_prep<<<(HDq * NRP2 + 255) / 256, 256, 0, stream>>>(Ev, EvT);

  qkv_gemm<<<dim3(16, 12, 3), 256, 0, stream>>>(Xh, Wh, bq, bk, bv, QH, KH, Vt);
  rel_gemm<<<192, 256, 0, stream>>>(QH, KH, EkH, EqH, relW);
  attn_kernel<<<1536, 256, 0, stream>>>(QH, KH, Vt, relW, arc, mask, EvT, ctxH);
  out_gemm<<<dim3(16, 12), 256, 0, stream>>>(ctxH, WoH, bo, (float*)d_out);
}
